// MatrixLSTMCell_60215441490304
// MI455X (gfx1250) — compile-verified, matrix operand order checked by reading, not run on hardware
//
#include <hip/hip_runtime.h>
#include <hip/hip_bf16.h>
#include <math.h>

#define DIM 768
#define NH  12
#define DH  64
#define B_  4
#define S_  1024
#define LOG2E 1.44269504088896340736f

typedef __attribute__((ext_vector_type(16))) __bf16       v16bf;
typedef __attribute__((ext_vector_type(8)))  float        v8f;
typedef __attribute__((ext_vector_type(4)))  unsigned int u32x4;

union Frag32B { v16bf v; u32x4 u[2]; };

// ---------------------------------------------------------------------------
// Kernel 1: gate projections ig/fg = concat(q,k,v) @ W.T + b  (24 outputs/(b,s))
// One block per (b,s); 24 waves, one per (gate,head) output.
// ---------------------------------------------------------------------------
__global__ void __launch_bounds__(768) gates_kernel(
    const float* __restrict__ q, const float* __restrict__ k,
    const float* __restrict__ v,
    const float* __restrict__ igw, const float* __restrict__ igb,
    const float* __restrict__ fgw, const float* __restrict__ fgb,
    float* __restrict__ igRaw, float* __restrict__ fgRaw) {
  const int bs   = blockIdx.x;            // 0..B*S-1
  const int b    = bs / S_;
  const int s    = bs % S_;
  const int w    = threadIdx.x >> 5;      // 0..23
  const int lane = threadIdx.x & 31;
  const int head = w % NH;
  const bool isF = (w >= NH);
  const float* W    = (isF ? fgw : igw) + (size_t)head * (3 * DIM);
  const float* qrow = q + (size_t)bs * DIM;
  const float* krow = k + (size_t)bs * DIM;
  const float* vrow = v + (size_t)bs * DIM;
  float acc = 0.f;
  for (int f = lane; f < DIM; f += 32) {
    acc += qrow[f] * W[f] + krow[f] * W[DIM + f] + vrow[f] * W[2 * DIM + f];
  }
  #pragma unroll
  for (int off = 16; off >= 1; off >>= 1) acc += __shfl_xor(acc, off, 32);
  if (lane == 0) {
    const float bias = (isF ? fgb : igb)[head];
    float* dst = isF ? fgRaw : igRaw;
    dst[((size_t)b * NH + head) * S_ + s] = acc + bias;
  }
}

// ---------------------------------------------------------------------------
// Kernel 2: per (b,h): c = cumsum(logsigmoid(fg));  a = ig - c;
//           M = prefixmax(a).
// Stores base-2 pre-scaled values so the main loop uses raw v_exp_f32:
//   gG2 = a*log2e,  gNM2 = -M*log2e,  gE = exp(-(c+M)).
// ---------------------------------------------------------------------------
__global__ void __launch_bounds__(1024) scan_kernel(
    const float* __restrict__ igRaw, const float* __restrict__ fgRaw,
    float* __restrict__ gG2, float* __restrict__ gNM2, float* __restrict__ gE) {
  __shared__ float buf[S_];
  const int t = threadIdx.x;
  const size_t base = (size_t)blockIdx.x * S_;
  const float fg = fgRaw[base + t];
  const float ls = fminf(fg, 0.f) - log1pf(__expf(-fabsf(fg)));
  buf[t] = ls;
  __syncthreads();
  for (int off = 1; off < S_; off <<= 1) {
    const float p = (t >= off) ? buf[t - off] : 0.f;
    __syncthreads();
    buf[t] += p;
    __syncthreads();
  }
  const float c = buf[t];
  const float a = igRaw[base + t] - c;
  __syncthreads();
  buf[t] = a;
  __syncthreads();
  for (int off = 1; off < S_; off <<= 1) {
    const float p = (t >= off) ? buf[t - off] : -INFINITY;
    __syncthreads();
    buf[t] = fmaxf(buf[t], p);
    __syncthreads();
  }
  const float M = buf[t];
  gG2[base + t]  = a * LOG2E;
  gNM2[base + t] = -M * LOG2E;
  gE[base + t]   = __expf(-(c + M));
}

// ---------------------------------------------------------------------------
// Kernel 3: fp32 -> bf16 repack. q scaled by 1/sqrt(DH).
//   qbf,kbf: (B,NH,S,DH)  vbfT: (B,NH,DH,S)
// ---------------------------------------------------------------------------
__global__ void __launch_bounds__(256) pack_kernel(
    const float* __restrict__ q, const float* __restrict__ k,
    const float* __restrict__ v,
    __bf16* __restrict__ qbf, __bf16* __restrict__ kbf,
    __bf16* __restrict__ vbfT) {
  const size_t idx = (size_t)blockIdx.x * blockDim.x + threadIdx.x;
  if (idx >= (size_t)B_ * S_ * DIM) return;
  const int dmm = (int)(idx % DIM);
  const size_t bs = idx / DIM;
  const int s = (int)(bs % S_);
  const int b = (int)(bs / S_);
  const int h = dmm / DH, d = dmm % DH;
  const size_t bh = (size_t)b * NH + h;
  qbf[(bh * S_ + s) * DH + d]  = (__bf16)(q[idx] * 0.125f);
  kbf[(bh * S_ + s) * DH + d]  = (__bf16)(k[idx]);
  vbfT[(bh * DH + d) * S_ + s] = (__bf16)(v[idx]);
}

// ---------------------------------------------------------------------------
// Kernel 4: main mLSTM. One wave per 32-row tile (two 16-row sub-tiles share
// all K/V B-fragments -> 16 WMMA per 32-key step). Causal mask applied only
// in the single tail step; interior steps are mask-free.
// Per score element: D = 2^(g2[j] + negM2[i])  (one v_add + one v_exp_f32).
// ---------------------------------------------------------------------------
#define MLSTM_STEP(J0, MASKED)                                                 \
  {                                                                            \
    const int j0_ = (J0);                                                      \
    _Pragma("unroll")                                                          \
    for (int t = 0; t < 2; ++t) {                                              \
      const int jb = j0_ + t * 16;                                             \
      Frag32B kb0, kb1;                                                        \
      const __bf16* krow = kbf + (bh * S_ + (size_t)(jb + lane16)) * DH;       \
      kb0.u[0] = *(const u32x4*)(krow + half * 8);                             \
      kb0.u[1] = *(const u32x4*)(krow + 16 + half * 8);                        \
      kb1.u[0] = *(const u32x4*)(krow + 32 + half * 8);                        \
      kb1.u[1] = *(const u32x4*)(krow + 48 + half * 8);                        \
      const int   jcol  = jb + lane16;                                         \
      const float gcol2 = gG2[bh * S_ + jcol];                                 \
      _Pragma("unroll")                                                        \
      for (int rt = 0; rt < 2; ++rt) {                                         \
        v8f acc = (v8f){0.f, 0.f, 0.f, 0.f, 0.f, 0.f, 0.f, 0.f};               \
        acc = __builtin_amdgcn_wmma_f32_16x16x32_bf16(                         \
            false, a0[rt].v, false, kb0.v, (short)0, acc, false, false);       \
        acc = __builtin_amdgcn_wmma_f32_16x16x32_bf16(                         \
            false, a1[rt].v, false, kb1.v, (short)0, acc, false, false);       \
        _Pragma("unroll")                                                      \
        for (int r = 0; r < 8; ++r) {                                          \
          float Ce = acc[r] * __builtin_amdgcn_exp2f(gcol2 + negM2[rt][r]);    \
          if (MASKED) {                                                        \
            const int mrow = i0 + rt * 16 + half * 8 + r;                      \
            Ce = (jcol <= mrow) ? Ce : 0.f;                                    \
          }                                                                    \
          rowsum[rt][r] += Ce;                                                 \
          Cbuf[(rt * 16 + half * 8 + r) * 40 + t * 16 + lane16] = (__bf16)Ce;  \
        }                                                                      \
      }                                                                        \
    }                                                                          \
    asm volatile("s_wait_dscnt 0" ::: "memory");                               \
    Frag32B ca0, ca1;                                                          \
    {                                                                          \
      const __bf16* crow0 = Cbuf + lane16 * 40;                                \
      ca0.u[0] = *(const u32x4*)(crow0 + half * 8);                            \
      ca0.u[1] = *(const u32x4*)(crow0 + 16 + half * 8);                       \
      const __bf16* crow1 = Cbuf + (16 + lane16) * 40;                         \
      ca1.u[0] = *(const u32x4*)(crow1 + half * 8);                            \
      ca1.u[1] = *(const u32x4*)(crow1 + 16 + half * 8);                       \
    }                                                                          \
    _Pragma("unroll")                                                          \
    for (int tt = 0; tt < 4; ++tt) {                                           \
      Frag32B vb;                                                              \
      const __bf16* vrow = vbfT + (bh * DH + (size_t)(tt * 16 + lane16)) * S_; \
      vb.u[0] = *(const u32x4*)(vrow + j0_ + half * 8);                        \
      vb.u[1] = *(const u32x4*)(vrow + j0_ + 16 + half * 8);                   \
      hacc0[tt] = __builtin_amdgcn_wmma_f32_16x16x32_bf16(                     \
          false, ca0.v, false, vb.v, (short)0, hacc0[tt], false, false);       \
      hacc1[tt] = __builtin_amdgcn_wmma_f32_16x16x32_bf16(                     \
          false, ca1.v, false, vb.v, (short)0, hacc1[tt], false, false);       \
    }                                                                          \
  }

__global__ void __launch_bounds__(32) mlstm_kernel(
    const __bf16* __restrict__ qbf, const __bf16* __restrict__ kbf,
    const __bf16* __restrict__ vbfT,
    const float* __restrict__ gG2, const float* __restrict__ gNM2,
    const float* __restrict__ gE,
    const float* __restrict__ norm_w, const float* __restrict__ norm_b,
    float* __restrict__ out) {
  __shared__ __bf16 Cbuf[32 * 40];       // 32 rows x 32 keys (+8 pad), bf16

  const int lane   = threadIdx.x & 31;
  const int lane16 = lane & 15;
  const int half   = lane >> 4;          // K-half / M-half selector
  const int i0     = blockIdx.x * 32;    // row tile base (32 rows)
  const int hh     = blockIdx.y;
  const int bb     = blockIdx.z;
  const size_t bh  = (size_t)bb * NH + hh;

  // --- Q A-fragments for both 16-row sub-tiles (ISA 16-bit A layout)
  Frag32B a0[2], a1[2];
  #pragma unroll
  for (int rt = 0; rt < 2; ++rt) {
    const __bf16* qrow =
        qbf + (bh * S_ + (size_t)(i0 + rt * 16 + lane16)) * DH;
    a0[rt].u[0] = *(const u32x4*)(qrow + half * 8);
    a0[rt].u[1] = *(const u32x4*)(qrow + 16 + half * 8);
    a1[rt].u[0] = *(const u32x4*)(qrow + 32 + half * 8);
    a1[rt].u[1] = *(const u32x4*)(qrow + 48 + half * 8);
  }

  // C/D layout rows this lane covers: M = rt*16 + half*8 + r
  float negM2[2][8];
  #pragma unroll
  for (int rt = 0; rt < 2; ++rt)
    #pragma unroll
    for (int r = 0; r < 8; ++r)
      negM2[rt][r] = gNM2[bh * S_ + i0 + rt * 16 + half * 8 + r];

  v8f hacc0[4], hacc1[4];
  #pragma unroll
  for (int t = 0; t < 4; ++t) {
    hacc0[t] = (v8f){0.f, 0.f, 0.f, 0.f, 0.f, 0.f, 0.f, 0.f};
    hacc1[t] = (v8f){0.f, 0.f, 0.f, 0.f, 0.f, 0.f, 0.f, 0.f};
  }
  float rowsum[2][8] = {{0.f}};

  // interior (mask-free) steps: cols j0..j0+31 all <= i0 (min row)
  const int nfull = i0 >> 5;
  for (int st = 0; st < nfull; ++st) {
    const int j0 = st * 32;
    __builtin_prefetch(kbf + (bh * S_ + (size_t)(j0 + 32 + lane16)) * DH, 0, 1);
    MLSTM_STEP(j0, 0)
  }
  // single triangular tail step: cols i0..i0+31 vs rows i0..i0+31
  MLSTM_STEP(i0, 1)

  // --- row sums across the 16 lanes holding each row (masks <16 stay in-half)
  #pragma unroll
  for (int rt = 0; rt < 2; ++rt)
    #pragma unroll
    for (int r = 0; r < 8; ++r) {
      #pragma unroll
      for (int off = 8; off >= 1; off >>= 1)
        rowsum[rt][r] += __shfl_xor(rowsum[rt][r], off, 32);
    }
  float inv[2][8];
  #pragma unroll
  for (int rt = 0; rt < 2; ++rt)
    #pragma unroll
    for (int r = 0; r < 8; ++r) {
      const float E = gE[bh * S_ + i0 + rt * 16 + half * 8 + r];
      inv[rt][r] = 1.f / (fmaxf(fabsf(rowsum[rt][r]), E) + 1e-6f);
    }

  // --- normalize + fused per-head layernorm over DH=64
  float wa[4], ba[4];
  #pragma unroll
  for (int tt = 0; tt < 4; ++tt) {
    const int col = hh * DH + tt * 16 + lane16;
    wa[tt] = 1.f + norm_w[col];
    ba[tt] = norm_b[col];
  }
  #pragma unroll
  for (int rt = 0; rt < 2; ++rt) {
    v8f* hacc = rt ? hacc1 : hacc0;
    float sum[8] = {0.f, 0.f, 0.f, 0.f, 0.f, 0.f, 0.f, 0.f};
    float sq[8]  = {0.f, 0.f, 0.f, 0.f, 0.f, 0.f, 0.f, 0.f};
    #pragma unroll
    for (int tt = 0; tt < 4; ++tt) {
      #pragma unroll
      for (int r = 0; r < 8; ++r) {
        const float h = hacc[tt][r] * inv[rt][r];
        hacc[tt][r] = h;
        sum[r] += h;
        sq[r]  += h * h;
      }
    }
    #pragma unroll
    for (int r = 0; r < 8; ++r) {
      #pragma unroll
      for (int off = 8; off >= 1; off >>= 1) {
        sum[r] += __shfl_xor(sum[r], off, 32);
        sq[r]  += __shfl_xor(sq[r],  off, 32);
      }
    }
    #pragma unroll
    for (int r = 0; r < 8; ++r) {
      const float mean = sum[r] * (1.f / 64.f);
      const float var  = sq[r] * (1.f / 64.f) - mean * mean;
      const float rstd = rsqrtf(var + 1e-5f);
      const int mrow = i0 + rt * 16 + half * 8 + r;
      float* orow = out + ((size_t)bb * S_ + mrow) * DIM + hh * DH;
      #pragma unroll
      for (int tt = 0; tt < 4; ++tt)
        orow[tt * 16 + lane16] = (hacc[tt][r] - mean) * rstd * wa[tt] + ba[tt];
    }
  }
}

// ---------------------------------------------------------------------------
extern "C" void kernel_launch(void* const* d_in, const int* in_sizes, int n_in,
                              void* d_out, int out_size, void* d_ws,
                              size_t ws_size, hipStream_t stream) {
  const float* q   = (const float*)d_in[0];
  const float* k   = (const float*)d_in[1];
  const float* v   = (const float*)d_in[2];
  const float* igw = (const float*)d_in[3];
  const float* igb = (const float*)d_in[4];
  const float* fgw = (const float*)d_in[5];
  const float* fgb = (const float*)d_in[6];
  const float* nw  = (const float*)d_in[7];
  const float* nb  = (const float*)d_in[8];
  float* out = (float*)d_out;

  char* ws = (char*)d_ws;
  size_t o = 0;
  auto alloc = [&](size_t bytes) -> void* {
    void* p = ws + o;
    o = (o + bytes + 255) & ~(size_t)255;
    return p;
  };
  const size_t nBHS = (size_t)B_ * NH * S_;       // 49152
  const size_t nQKV = (size_t)B_ * NH * S_ * DH;  // 3145728

  __bf16* qbf   = (__bf16*)alloc(nQKV * sizeof(__bf16));
  __bf16* kbf   = (__bf16*)alloc(nQKV * sizeof(__bf16));
  __bf16* vbfT  = (__bf16*)alloc(nQKV * sizeof(__bf16));
  float*  igRaw = (float*)alloc(nBHS * sizeof(float));
  float*  fgRaw = (float*)alloc(nBHS * sizeof(float));
  float*  gG2   = (float*)alloc(nBHS * sizeof(float));
  float*  gNM2  = (float*)alloc(nBHS * sizeof(float));
  float*  gE    = (float*)alloc(nBHS * sizeof(float));

  gates_kernel<<<B_ * S_, 768, 0, stream>>>(q, k, v, igw, igb, fgw, fgb,
                                            igRaw, fgRaw);
  scan_kernel<<<B_ * NH, S_, 0, stream>>>(igRaw, fgRaw, gG2, gNM2, gE);
  const size_t nElem = (size_t)B_ * S_ * DIM;
  pack_kernel<<<(unsigned)((nElem + 255) / 256), 256, 0, stream>>>(
      q, k, v, qbf, kbf, vbfT);
  mlstm_kernel<<<dim3(S_ / 32, NH, B_), 32, 0, stream>>>(
      qbf, kbf, vbfT, gG2, gNM2, gE, nw, nb, out);
}
